// SSIMLoss_50921132261556
// MI455X (gfx1250) — matrix operand order repaired and checked by reading, not compiled, not run on hardware
//
#include <hip/hip_runtime.h>
#include <math.h>

typedef __attribute__((ext_vector_type(16))) _Float16 v16h;
typedef __attribute__((ext_vector_type(8)))  _Float16 v8h;
typedef __attribute__((ext_vector_type(8)))  float    v8f;
typedef __attribute__((ext_vector_type(4)))  float    v4f;

#define IMG 512
#define PLANES 48                       // 16 * 3
#define STRIP_ROWS 64
#define STRIP_COLS 16
#define STRIPS_PER_PLANE 256            // (512/64) * (512/16)
#define NSTRIPS (PLANES * STRIPS_PER_PLANE)   // 12288
#define WAVES_PER_BLOCK 4
#define NBLOCKS (NSTRIPS / WAVES_PER_BLOCK)   // 3072
#define USTRIDE 88                      // halves per H-column (80 rows + pad, 176B = 16B aligned)
#define QSZ (16 * USTRIDE)              // halves per quantity buffer
#define WAVESZ (5 * QSZ)                // halves per wave

// band(d) = gaussian tap g[d] for d in [0,10], else 0 (register-resident select chain)
__device__ __forceinline__ float bandf(int d, const float g[11]) {
    float r = 0.f;
#pragma unroll
    for (int k = 0; k < 11; ++k) r = (d == k) ? g[k] : r;
    return r;
}

__global__ __launch_bounds__(128) void ssim_main(const float* __restrict__ pred,
                                                 const float* __restrict__ targ,
                                                 const float* __restrict__ win,
                                                 float* __restrict__ partial) {
    __shared__ __align__(16) _Float16 HB[WAVES_PER_BLOCK * WAVESZ];
    __shared__ float red[WAVES_PER_BLOCK];

    const int tid  = threadIdx.x;
    const int lane = tid & 31;
    const int wid  = tid >> 5;
    const int hi   = (lane >> 4) & 1;
    const int l15  = lane & 15;

    const int S     = blockIdx.x * WAVES_PER_BLOCK + wid;
    const int plane = S / STRIPS_PER_PLANE;
    const int t     = S - plane * STRIPS_PER_PLANE;
    const int sr    = t >> 5;           // 0..7
    const int sc    = t & 31;           // 0..31
    const int r0    = sr * STRIP_ROWS;
    const int c0    = sc * STRIP_COLS;

    const float* P = pred + (size_t)plane * IMG * IMG;
    const float* T = targ + (size_t)plane * IMG * IMG;

    // Derive separable gaussian taps from the 2D window input: g[j] = k2d[5][j] / sqrt(k2d[5][5])
    float g[11];
    {
        float s = sqrtf(win[5 * 11 + 5]);
#pragma unroll
        for (int j = 0; j < 11; ++j) g[j] = win[5 * 11 + j] / s;
    }

    // Constant banded matrices (f16).
    // Av (vertical, A-layout 16x32):  Av[i][K] = band(K - i)
    // Bh (horizontal, B-layout 32x16): Bh[K][n] = band(K - n - 3)
    v16h Av, Bh;
#pragma unroll
    for (int e = 0; e < 16; ++e) {
        int Ka = e + (hi ? 16 : 0);                       // contiguous K index (16*hi + e), same order as vertical-pass B
        Av[e] = (_Float16)bandf(Ka - l15, g);
        int Kb = e + ((e >= 8) ? 8 : 0) + (hi ? 8 : 0);   // 16-bit B-layout K index
        Bh[e] = (_Float16)bandf(Kb - l15 - 3, g);
    }

    _Float16* myH = HB + wid * WAVESZ;
    const bool colInterior = (c0 >= 8) && (c0 + 32 <= IMG);
    const int  cbase = c0 - 8;

    // ---------------- Horizontal pass: 5 row-tiles of 16 rows ----------------
    for (int a = 0; a < 5; ++a) {
        const int row = r0 - 5 + 16 * a + l15;            // same row for both lane halves
        float p[16], q[16];
#pragma unroll
        for (int e = 0; e < 16; ++e) { p[e] = 0.f; q[e] = 0.f; }
        if ((unsigned)row < IMG) {
            const float* Pr = P + row * IMG;
            const float* Tr = T + row * IMG;
            if (colInterior) {
                const int cA = cbase + hi * 8;            // 8-aligned
                v4f p0 = *(const v4f*)(Pr + cA);
                v4f p1 = *(const v4f*)(Pr + cA + 4);
                v4f p2 = *(const v4f*)(Pr + cA + 16);
                v4f p3 = *(const v4f*)(Pr + cA + 20);
                v4f q0 = *(const v4f*)(Tr + cA);
                v4f q1 = *(const v4f*)(Tr + cA + 4);
                v4f q2 = *(const v4f*)(Tr + cA + 16);
                v4f q3 = *(const v4f*)(Tr + cA + 20);
#pragma unroll
                for (int k = 0; k < 4; ++k) {
                    p[k] = p0[k]; p[4 + k] = p1[k]; p[8 + k] = p2[k]; p[12 + k] = p3[k];
                    q[k] = q0[k]; q[4 + k] = q1[k]; q[8 + k] = q2[k]; q[12 + k] = q3[k];
                }
            } else {
#pragma unroll
                for (int e = 0; e < 16; ++e) {
                    int c = cbase + (e & 7) + ((e >= 8) ? 16 : 0) + hi * 8;
                    if ((unsigned)c < IMG) { p[e] = Pr[c]; q[e] = Tr[c]; }
                }
            }
        }
        // f16 operands for the 5 quantities
        v16h ph, th;
#pragma unroll
        for (int e = 0; e < 16; ++e) { ph[e] = (_Float16)p[e]; th[e] = (_Float16)q[e]; }
        v16h pp = ph * ph, tt = th * th, pt = ph * th;

        v8f z = {};
        v8f h0 = __builtin_amdgcn_wmma_f32_16x16x32_f16(false, ph, false, Bh, (short)0, z, false, false);
        v8f h1 = __builtin_amdgcn_wmma_f32_16x16x32_f16(false, th, false, Bh, (short)0, z, false, false);
        v8f h2 = __builtin_amdgcn_wmma_f32_16x16x32_f16(false, pp, false, Bh, (short)0, z, false, false);
        v8f h3 = __builtin_amdgcn_wmma_f32_16x16x32_f16(false, tt, false, Bh, (short)0, z, false, false);
        v8f h4 = __builtin_amdgcn_wmma_f32_16x16x32_f16(false, pt, false, Bh, (short)0, z, false, false);

        const int u = 16 * a + hi * 8;                    // strip row of this lane's 8 results
        auto storeH = [&](int qi, const v8f& acc) {
            v8h hh;
#pragma unroll
            for (int k = 0; k < 8; ++k) hh[k] = (_Float16)acc[k];
            *(v8h*)(myH + qi * QSZ + l15 * USTRIDE + u) = hh;   // one ds_store_b128
        };
        storeH(0, h0); storeH(1, h1); storeH(2, h2); storeH(3, h3); storeH(4, h4);
    }

    // ---------------- Vertical pass + SSIM epilogue: 4 output tiles ----------------
    const float C1 = 0.0001f, C2 = 0.0009f;
    float lsum = 0.f;
    for (int s = 0; s < 4; ++s) {
        v8f acc[5];
#pragma unroll
        for (int qi = 0; qi < 5; ++qi) {
            const int u = 16 * s + hi * 16;
            const v8h* src = (const v8h*)(myH + qi * QSZ + l15 * USTRIDE + u);
            v8h b0 = src[0], b1 = src[1];                 // two ds_load_b128, already f16
            v16h B;
#pragma unroll
            for (int k = 0; k < 8; ++k) { B[k] = b0[k]; B[8 + k] = b1[k]; }
            v8f z = {};
            acc[qi] = __builtin_amdgcn_wmma_f32_16x16x32_f16(false, Av, false, B, (short)0, z, false, false);
        }
#pragma unroll
        for (int k = 0; k < 8; ++k) {
            float mu1 = acc[0][k], mu2 = acc[1][k];
            float mu1sq = mu1 * mu1, mu2sq = mu2 * mu2, mu12 = mu1 * mu2;
            float s1  = acc[2][k] - mu1sq;
            float s2  = acc[3][k] - mu2sq;
            float s12 = acc[4][k] - mu12;
            float num = (2.f * mu12 + C1) * (2.f * s12 + C2);
            float den = (mu1sq + mu2sq + C1) * (s1 + s2 + C2);
            lsum += num / den;
        }
    }

    // wave + block reduction -> one partial per block
#pragma unroll
    for (int off = 16; off > 0; off >>= 1) lsum += __shfl_xor(lsum, off, 32);
    if (lane == 0) red[wid] = lsum;
    __syncthreads();
    if (tid == 0) partial[blockIdx.x] = red[0] + red[1] + red[2] + red[3];
}

__global__ __launch_bounds__(256) void ssim_finish(const float* __restrict__ partial,
                                                   float* __restrict__ out, int n) {
    __shared__ float red[256];
    float s = 0.f;
    for (int i = threadIdx.x; i < n; i += 256) s += partial[i];
    red[threadIdx.x] = s;
    __syncthreads();
    for (int k = 128; k > 0; k >>= 1) {
        if ((int)threadIdx.x < k) red[threadIdx.x] += red[threadIdx.x + k];
        __syncthreads();
    }
    if (threadIdx.x == 0) {
        const float N = 16.f * 3.f * 512.f * 512.f;
        out[0] = 0.1f * (1.f - red[0] / N);
    }
}

extern "C" void kernel_launch(void* const* d_in, const int* in_sizes, int n_in,
                              void* d_out, int out_size, void* d_ws, size_t ws_size,
                              hipStream_t stream) {
    (void)in_sizes; (void)n_in; (void)out_size; (void)ws_size;
    const float* pred = (const float*)d_in[0];
    const float* targ = (const float*)d_in[1];
    const float* win  = (const float*)d_in[2];
    float* partial = (float*)d_ws;                         // NBLOCKS floats of scratch
    ssim_main<<<NBLOCKS, 128, 0, stream>>>(pred, targ, win, partial);
    ssim_finish<<<1, 256, 0, stream>>>(partial, (float*)d_out, NBLOCKS);
}
